// FFT1D_block_23536420782231
// MI455X (gfx1250) — matrix operand order repaired and checked by reading, not compiled, not run on hardware
//
#include <hip/hip_runtime.h>
#include <hip/hip_bf16.h>
#include <stdint.h>

// ---------------------------------------------------------------------------
// FNO spectral conv layer == single real 512x512 channel-mixing matrix M:
//   out[b,c,x] = sum_i M[c,i] * x[b,i,x]
// Precompute M (f32 DFT passes -> bf16), then batched bf16-WMMA GEMM.
// Roofline: 537MB HBM min traffic -> ~23us @ 23.3 TB/s; bf16 WMMA keeps the
// 68.7 GFLOP compute under that, so the kernel is engineered around one-pass
// streaming of x/out (non-temporal) with M served from L2 (192MB).
// ---------------------------------------------------------------------------

typedef __attribute__((ext_vector_type(16))) __bf16 v16bf;
typedef __attribute__((ext_vector_type(8)))  __bf16 bf16x8;
typedef __attribute__((ext_vector_type(8)))  float  v8f;
typedef __attribute__((ext_vector_type(4)))  float  v4f;   // true clang vector
                                                           // (NT builtins OK)

#define C_IN  512
#define C_OUT 512
#define XDIM  4096
#define BATCH 32
#define BK    32      // K step (== one WMMA K)
#define BN    64      // x-columns per block
#define LDB   40      // xs row stride in bf16 (32 + 8 pad) -> 80B, 16B aligned

// ---- twiddle tables: angle depends only on (o*c mod 512) --------------------
__global__ __launch_bounds__(512) void build_tables_k(float* ct, float* st) {
  int i = threadIdx.x;
  double ang = 6.283185307179586476925286766559 * (double)i / 512.0;
  ct[i] = (float)cos(ang);
  st[i] = (float)sin(ang);
}

// ---- stage 1: S[k,c] = sum_o (wr+j*wi)[k,o] * e^{+2pi j * o c / 512} --------
__global__ __launch_bounds__(256) void stage1_k(const float* __restrict__ wr,
                                                const float* __restrict__ wi,
                                                const float* __restrict__ ctg,
                                                const float* __restrict__ stg,
                                                float* __restrict__ P,
                                                float* __restrict__ Q) {
  __shared__ float ct[512], st[512];
  for (int i = threadIdx.x; i < 512; i += 256) { ct[i] = ctg[i]; st[i] = stg[i]; }
  __syncthreads();
  int g = blockIdx.x * 256 + threadIdx.x;
  int k = g >> 9;                            // uniform within block
  int c = g & 511;
  float p = 0.f, q = 0.f;
  for (int o = 0; o < 512; ++o) {
    int idx = (o * c) & 511;
    float cv = ct[idx], sv = st[idx];
    float a = wr[k * 512 + o];               // wave-uniform -> scalar load
    float bb = wi[k * 512 + o];
    p += a * cv - bb * sv;
    q += a * sv + bb * cv;
  }
  P[k * 512 + c] = p;
  Q[k * 512 + c] = q;
}

// ---- stage 2: M[c,i] = (1/512) sum_k P[k,c]cos(2pi ki/512)+Q[k,c]sin(...) ---
__global__ __launch_bounds__(256) void stage2_k(const float* __restrict__ P,
                                                const float* __restrict__ Q,
                                                const float* __restrict__ ctg,
                                                const float* __restrict__ stg,
                                                __bf16* __restrict__ Mb) {
  __shared__ float ct[512], st[512];
  for (int i = threadIdx.x; i < 512; i += 256) { ct[i] = ctg[i]; st[i] = stg[i]; }
  __syncthreads();
  int g = blockIdx.x * 256 + threadIdx.x;
  int c = g >> 9;                            // uniform within block
  int ii = g & 511;
  float m = 0.f;
  for (int k = 0; k < 512; ++k) {
    int idx = (k * ii) & 511;
    m += P[k * 512 + c] * ct[idx] + Q[k * 512 + c] * st[idx];
  }
  Mb[c * 512 + ii] = (__bf16)(m * (1.0f / 512.0f));  // row-major A matrix
}

// ---- fragment gather: two 16B loads (works for LDS or global pointers) ------
template <typename T>
static __device__ inline v16bf ld_frag(const T* p0, const T* p1) {
  union { v16bf v; bf16x8 h[2]; } u;
  u.h[0] = *(const bf16x8*)p0;
  u.h[1] = *(const bf16x8*)p1;
  return u.v;
}

// ---- main GEMM: out[b] = M(512x512,bf16) * x[b](512x4096,f32->bf16) ---------
// Block: 256 threads = 8 wave32; block tile 512(M) x 64(N); wave tile 64x64.
// * A fragments gathered straight from global (M is L2-resident, 512KB).
// * x tile (needs f32->bf16 transpose-convert) staged through double-buffered
//   LDS; next tile's HBM loads issued before this tile's barrier (pipelined).
// * x loads / out stores non-temporal so streams don't evict M from L2.
__global__ __launch_bounds__(256) void fno_gemm_k(const __bf16* __restrict__ Mb,
                                                  const float* __restrict__ Xg,
                                                  float* __restrict__ Out) {
  __shared__ __bf16 xs[2][BN * LDB];   // 2 x 5120 B ping-pong, xs[n][k]
  const int t    = threadIdx.x;
  const int b    = blockIdx.z;
  const int xn0  = blockIdx.x * BN;
  const int wave = t >> 5;
  const int lane = t & 31;
  const int lo   = lane & 15;
  const int hi   = lane >> 4;
  const int m0w  = wave * 64;          // 8 waves cover all 512 output rows

  v8f acc[4][4];
#pragma unroll
  for (int mi = 0; mi < 4; ++mi)
#pragma unroll
    for (int ni = 0; ni < 4; ++ni)
      acc[mi][ni] = (v8f){0.f, 0.f, 0.f, 0.f, 0.f, 0.f, 0.f, 0.f};

  // x-tile staging: thread owns k-pair (kk,kk+1) x 4 consecutive n.
  const int kk = (t >> 4) * 2;         // 0,2,..,30
  const int nq = (t & 15) * 4;
  const v4f* xbase = (const v4f*)(Xg + (size_t)b * C_IN * XDIM + xn0 + nq);
  // prologue: issue kt=0 loads (non-temporal: x is read exactly once)
  v4f a0 = __builtin_nontemporal_load(xbase + (size_t)kk * (XDIM / 4));
  v4f a1 = __builtin_nontemporal_load(xbase + (size_t)(kk + 1) * (XDIM / 4));

  for (int kt = 0; kt < C_IN / BK; ++kt) {
    const int k0 = kt * BK;
    // --- transpose-convert this tile into LDS ping-pong buffer ---
    uint32_t* dst = (uint32_t*)xs[kt & 1];
#pragma unroll
    for (int j = 0; j < 4; ++j) {
      union { __bf16 h[2]; uint32_t u; } pk;
      pk.h[0] = (__bf16)a0[j];       // k even
      pk.h[1] = (__bf16)a1[j];       // k odd (contiguous in xs[n][k])
      dst[(nq + j) * (LDB / 2) + (kk >> 1)] = pk.u;
    }
    // --- issue next tile's HBM loads now (overlap with barrier + WMMA) ---
    if (kt + 1 < C_IN / BK) {
      a0 = __builtin_nontemporal_load(xbase + (size_t)(k0 + BK + kk) * (XDIM / 4));
      a1 = __builtin_nontemporal_load(xbase + (size_t)(k0 + BK + kk + 1) * (XDIM / 4));
    }
    // --- A fragments straight from global (L2-resident M), before barrier ---
    // A 16x32 layout: row = lo; hi=0 -> K 0..7 & 16..23, hi=1 -> K 8..15 & 24..31
    v16bf afr[4];
#pragma unroll
    for (int mi = 0; mi < 4; ++mi) {
      const __bf16* rp = Mb + (size_t)(m0w + mi * 16 + lo) * C_IN + k0 + 8 * hi;
      afr[mi] = ld_frag(rp, rp + 16);
    }
    __syncthreads();   // one barrier/iter; ping-pong makes the second redundant
    // --- B fragments from LDS: col = lo; hi=0 -> K 0..7 & 16..23, hi=1 -> K 8..15 & 24..31 ---
    v16bf bfr[4];
#pragma unroll
    for (int ni = 0; ni < 4; ++ni) {
      const __bf16* cp = &xs[kt & 1][(ni * 16 + lo) * LDB + 8 * hi];
      bfr[ni] = ld_frag(cp, cp + 16);
    }
#pragma unroll
    for (int mi = 0; mi < 4; ++mi)
#pragma unroll
      for (int ni = 0; ni < 4; ++ni)
        acc[mi][ni] = __builtin_amdgcn_wmma_f32_16x16x32_bf16(
            false, afr[mi], false, bfr[ni], (short)0, acc[mi][ni], false, false);
  }

  // --- store C (VGPR j holds row 8*hi + j, col = lo); non-temporal stream ---
#pragma unroll
  for (int mi = 0; mi < 4; ++mi) {
#pragma unroll
    for (int ni = 0; ni < 4; ++ni) {
      int row = m0w + mi * 16 + hi * 8;
      float* op = Out + ((size_t)b * C_OUT + row) * XDIM + xn0 + ni * 16 + lo;
#pragma unroll
      for (int j = 0; j < 8; ++j)
        __builtin_nontemporal_store(acc[mi][ni][j], op + (size_t)j * XDIM);
    }
  }
}

extern "C" void kernel_launch(void* const* d_in, const int* in_sizes, int n_in,
                              void* d_out, int out_size, void* d_ws, size_t ws_size,
                              hipStream_t stream) {
  (void)in_sizes; (void)n_in; (void)out_size; (void)ws_size;
  const float* x  = (const float*)d_in[0];   // [32, 512, 4096]
  const float* wr = (const float*)d_in[1];   // [512, 512]
  const float* wi = (const float*)d_in[2];   // [512, 512]
  float* out = (float*)d_out;                // [32, 512, 4096]

  // workspace layout (rebuilt every call -> deterministic):
  //   cos[512] | sin[512] | P[512*512] | Q[512*512] | M_bf16[512*512]
  float* ct = (float*)d_ws;
  float* st = ct + 512;
  float* P  = st + 512;
  float* Q  = P + 512 * 512;
  __bf16* Mb = (__bf16*)(Q + 512 * 512);

  build_tables_k<<<1, 512, 0, stream>>>(ct, st);
  stage1_k<<<(512 * 512) / 256, 256, 0, stream>>>(wr, wi, ct, st, P, Q);
  stage2_k<<<(512 * 512) / 256, 256, 0, stream>>>(P, Q, ct, st, Mb);

  dim3 grid(XDIM / BN, 1, BATCH);  // 64 x-windows * 32 batches = 2048 blocks
  fno_gemm_k<<<grid, 256, 0, stream>>>(Mb, x, out);
}
